// AttentionalPropagation_88210038325764
// MI455X (gfx1250) — matrix operand order repaired and checked by reading, not compiled, not run on hardware
//
#include <hip/hip_runtime.h>

#define Bb 2
#define Hh 4
#define Nn 4096
#define Mm 4096
#define Dd 256

typedef __attribute__((ext_vector_type(16))) _Float16 v16h;
typedef __attribute__((ext_vector_type(8)))  _Float16 v8h;
typedef __attribute__((ext_vector_type(8)))  float    v8f;

// ---- DPP row-rotate reductions over a 16-lane row (pure VALU, no LDS) ----
template<int CTRL>
__device__ __forceinline__ float dppRor(float x) {
  return __builtin_bit_cast(float, __builtin_amdgcn_update_dpp(
      0, __builtin_bit_cast(int, x), CTRL, 0xf, 0xf, true));
}
__device__ __forceinline__ float rowMax16(float x) {
  x = fmaxf(x, dppRor<0x121>(x));   // row_ror:1
  x = fmaxf(x, dppRor<0x122>(x));   // row_ror:2
  x = fmaxf(x, dppRor<0x124>(x));   // row_ror:4
  x = fmaxf(x, dppRor<0x128>(x));   // row_ror:8
  return x;
}
__device__ __forceinline__ float rowSum16(float x) {
  x += dppRor<0x121>(x);
  x += dppRor<0x122>(x);
  x += dppRor<0x124>(x);
  x += dppRor<0x128>(x);
  return x;
}

// ---------------- projection: two pointwise convs (D->32->256), fp32 in, f16 out
// mode 0: Q  -> Qt[b][h][n][d]  (scaled by 0.125)
// mode 1: K  -> Kt[b][h][m][d]
// mode 2: V  -> Vm[b][h][d][m]
__global__ __launch_bounds__(256) void proj_kernel(
    const float* __restrict__ X, const float* __restrict__ W1,
    const float* __restrict__ b1, const float* __restrict__ W2,
    const float* __restrict__ b2, _Float16* __restrict__ Out,
    int L, int mode)
{
  __shared__ float sX[256*64];     // [c][n], stride 64 -> conflict free
  __shared__ float sW1[32*256];
  __shared__ float sW2[256*32];
  __shared__ float sH[32*64];      // hidden [i][n]
  const int t = threadIdx.x;
  const int b = blockIdx.y;
  const int n0 = blockIdx.x * 64;
  const float* Xb = X + (size_t)b*256*L;
  for (int e=t; e<32*256; e+=256) sW1[e]=W1[e];
  for (int e=t; e<256*32; e+=256) sW2[e]=W2[e];
  for (int e=t; e<256*64; e+=256) sX[e] = Xb[(size_t)(e>>6)*L + n0 + (e&63)];
  __syncthreads();
  const int n = t & 63, g = t >> 6;
  for (int ii=0; ii<8; ++ii) {
    int i = g*8+ii;
    float acc = b1[i];
    const float* w = &sW1[i*256];
    #pragma unroll 8
    for (int c=0;c<256;++c) acc = fmaf(w[c], sX[c*64+n], acc);
    sH[i*64+n] = acc;
  }
  __syncthreads();
  if (mode == 2) {
    for (int oo=0; oo<64; ++oo) {
      int o = g*64+oo;
      float acc = b2[o];
      const float* w = &sW2[o*32];
      #pragma unroll
      for (int i=0;i<32;++i) acc = fmaf(w[i], sH[i*64+n], acc);
      int h2 = o & 3, d = o >> 2;           // channel c = d*H + h
      Out[(((size_t)b*Hh + h2)*64 + d)*L + n0 + n] = (_Float16)acc;
    }
  } else {
    const float sc = (mode==0) ? 0.125f : 1.0f;   // fold 1/sqrt(DIM) into Q
    for (int h2=0; h2<4; ++h2) {
      union { _Float16 e[16]; v8h v[2]; } tmp;
      for (int dd=0; dd<16; ++dd) {
        int o = (g*16+dd)*4 + h2;
        float acc = b2[o];
        const float* w = &sW2[o*32];
        #pragma unroll
        for (int i=0;i<32;++i) acc = fmaf(w[i], sH[i*64+n], acc);
        tmp.e[dd] = (_Float16)(acc*sc);
      }
      _Float16* dp = Out + (((size_t)b*Hh + h2)*L + n0 + n)*64 + g*16;
      *(v8h*)dp       = tmp.v[0];
      *(v8h*)(dp + 8) = tmp.v[1];
    }
  }
}

// ---------------- flash attention: one wave per 16-row n-tile, 64-col m-blocks
__global__ __launch_bounds__(128) void attn_kernel(
    const _Float16* __restrict__ Qt, const _Float16* __restrict__ Kt,
    const _Float16* __restrict__ Vm, float* __restrict__ msgT,
    int N, int M)
{
  __shared__ __align__(16) _Float16 sP[4][16][72];   // per-wave P staging, padded rows
  const int lane = threadIdx.x & 31;
  const int wv   = threadIdx.x >> 5;
  const int h = blockIdx.y, b = blockIdx.z;
  const int n0 = (blockIdx.x*4 + wv)*16;
  const int half = lane >> 4;   // 0: lanes 0-15, 1: lanes 16-31
  const int lid  = lane & 15;

  const _Float16* Qb = Qt + ((size_t)b*Hh + h)*(size_t)N*64;
  const _Float16* Kb = Kt + ((size_t)b*Hh + h)*(size_t)M*64;
  const _Float16* Vb = Vm + ((size_t)b*Hh + h)*(size_t)64*M;

  // A-operand for Q: lane-half holds K(d) 0..15 / 16..31
  v16h aq0, aq1;
  {
    const _Float16* qrow = Qb + (size_t)(n0 + lid)*64 + half*16;
    union { v16h v; v8h p[2]; } u;
    u.p[0] = *(const v8h*)(qrow);      u.p[1] = *(const v8h*)(qrow + 8);
    aq0 = u.v;
    u.p[0] = *(const v8h*)(qrow + 32); u.p[1] = *(const v8h*)(qrow + 40);
    aq1 = u.v;
  }

  float rmax[8], rsum[8];              // rsum = per-lane partial (reduced at end)
  v8f O0 = {}, O1 = {}, O2 = {}, O3 = {};
  #pragma unroll
  for (int r=0;r<8;++r) { rmax[r] = -1e30f; rsum[r] = 0.0f; }

  for (int m0 = 0; m0 < M; m0 += 64) {
    // ---- S = Q^T K : four 16-col subtiles, two d k-steps each (8 WMMAs)
    v8f s0 = {}, s1 = {}, s2 = {}, s3 = {};
    {
      const _Float16* kp = Kb + (size_t)(m0 + lid)*64 + half*16;
      v16h k00 = *(const v16h*)(kp);
      v16h k01 = *(const v16h*)(kp + 32);
      s0 = __builtin_amdgcn_wmma_f32_16x16x32_f16(false, aq0, false, k00, (short)0, s0, false, false);
      s0 = __builtin_amdgcn_wmma_f32_16x16x32_f16(false, aq1, false, k01, (short)0, s0, false, false);
      v16h k10 = *(const v16h*)(kp + 16*64);
      v16h k11 = *(const v16h*)(kp + 16*64 + 32);
      s1 = __builtin_amdgcn_wmma_f32_16x16x32_f16(false, aq0, false, k10, (short)0, s1, false, false);
      s1 = __builtin_amdgcn_wmma_f32_16x16x32_f16(false, aq1, false, k11, (short)0, s1, false, false);
      v16h k20 = *(const v16h*)(kp + 32*64);
      v16h k21 = *(const v16h*)(kp + 32*64 + 32);
      s2 = __builtin_amdgcn_wmma_f32_16x16x32_f16(false, aq0, false, k20, (short)0, s2, false, false);
      s2 = __builtin_amdgcn_wmma_f32_16x16x32_f16(false, aq1, false, k21, (short)0, s2, false, false);
      v16h k30 = *(const v16h*)(kp + 48*64);
      v16h k31 = *(const v16h*)(kp + 48*64 + 32);
      s3 = __builtin_amdgcn_wmma_f32_16x16x32_f16(false, aq0, false, k30, (short)0, s3, false, false);
      s3 = __builtin_amdgcn_wmma_f32_16x16x32_f16(false, aq1, false, k31, (short)0, s3, false, false);
      if (m0 + 64 < M) {
        __builtin_prefetch(kp + 64*64, 0, 3);
        __builtin_prefetch(Vb + (size_t)lid*M + m0 + 64, 0, 3);
      }
    }
    // ---- online softmax: row max via DPP rotates, sum kept as per-lane partials
    #pragma unroll
    for (int r=0;r<8;++r) {
      float a0 = s0[r], a1 = s1[r], a2 = s2[r], a3 = s3[r];
      float mc = fmaxf(fmaxf(a0, a1), fmaxf(a2, a3));
      mc = rowMax16(mc);
      float mn = fmaxf(rmax[r], mc);
      float corr = __expf(rmax[r] - mn);
      rmax[r] = mn;
      float p0 = __expf(a0 - mn), p1 = __expf(a1 - mn);
      float p2 = __expf(a2 - mn), p3 = __expf(a3 - mn);
      rsum[r] = rsum[r]*corr + ((p0 + p1) + (p2 + p3));
      O0[r]*=corr; O1[r]*=corr; O2[r]*=corr; O3[r]*=corr;
      int row = r + half*8;
      sP[wv][row][lid]      = (_Float16)p0;
      sP[wv][row][16 + lid] = (_Float16)p1;
      sP[wv][row][32 + lid] = (_Float16)p2;
      sP[wv][row][48 + lid] = (_Float16)p3;
    }
    // ---- reload P in A-operand layout (intra-wave LDS is in-order)
    v16h ap0, ap1;
    {
      const _Float16* pr = &sP[wv][lid][half*16];
      union { v16h v; v8h p[2]; } u;
      u.p[0] = *(const v8h*)(pr);      u.p[1] = *(const v8h*)(pr + 8);
      ap0 = u.v;
      u.p[0] = *(const v8h*)(pr + 32); u.p[1] = *(const v8h*)(pr + 40);
      ap1 = u.v;
    }
    // ---- O += P x V : 4 d-column tiles x 2 m k-steps (8 WMMAs)
    {
      const _Float16* vp = Vb + (size_t)lid*M + m0 + half*16;
      v16h b00 = *(const v16h*)(vp);
      v16h b01 = *(const v16h*)(vp + 32);
      O0 = __builtin_amdgcn_wmma_f32_16x16x32_f16(false, ap0, false, b00, (short)0, O0, false, false);
      O0 = __builtin_amdgcn_wmma_f32_16x16x32_f16(false, ap1, false, b01, (short)0, O0, false, false);
      v16h b10 = *(const v16h*)(vp + (size_t)16*M);
      v16h b11 = *(const v16h*)(vp + (size_t)16*M + 32);
      O1 = __builtin_amdgcn_wmma_f32_16x16x32_f16(false, ap0, false, b10, (short)0, O1, false, false);
      O1 = __builtin_amdgcn_wmma_f32_16x16x32_f16(false, ap1, false, b11, (short)0, O1, false, false);
      v16h b20 = *(const v16h*)(vp + (size_t)32*M);
      v16h b21 = *(const v16h*)(vp + (size_t)32*M + 32);
      O2 = __builtin_amdgcn_wmma_f32_16x16x32_f16(false, ap0, false, b20, (short)0, O2, false, false);
      O2 = __builtin_amdgcn_wmma_f32_16x16x32_f16(false, ap1, false, b21, (short)0, O2, false, false);
      v16h b30 = *(const v16h*)(vp + (size_t)48*M);
      v16h b31 = *(const v16h*)(vp + (size_t)48*M + 32);
      O3 = __builtin_amdgcn_wmma_f32_16x16x32_f16(false, ap0, false, b30, (short)0, O3, false, false);
      O3 = __builtin_amdgcn_wmma_f32_16x16x32_f16(false, ap1, false, b31, (short)0, O3, false, false);
    }
  }
  // ---- reduce deferred sums, normalize, write head-major msgT[b][n][h*64+d]
  float* mb = msgT + (size_t)b*N*Dd;
  #pragma unroll
  for (int r=0;r<8;++r) {
    float inv = 1.0f / rowSum16(rsum[r]);
    int n = n0 + r + half*8;
    float* rowp = mb + (size_t)n*Dd + h*64;   // 16 lanes -> 64B contiguous
    rowp[lid]      = O0[r]*inv;
    rowp[16 + lid] = O1[r]*inv;
    rowp[32 + lid] = O2[r]*inv;
    rowp[48 + lid] = O3[r]*inv;
  }
}

// ---------------- merge: msg (n-major, head-major channels) -> two convs -> m2T
__global__ __launch_bounds__(256) void merge_kernel(
    const float* __restrict__ msgT, const float* __restrict__ W1,
    const float* __restrict__ b1, const float* __restrict__ W2,
    const float* __restrict__ b2, float* __restrict__ m2T, int N)
{
  __shared__ float sM[64][257];
  __shared__ float sW1[32*256];
  __shared__ float sW2[256*32];
  __shared__ float sH[64][33];
  const int t = threadIdx.x;
  const int b = blockIdx.y;
  const int n0 = blockIdx.x*64;
  const float* srcp = msgT + ((size_t)b*N + n0)*256;
  for (int e=t;e<32*256;e+=256) sW1[e]=W1[e];
  for (int e=t;e<256*32;e+=256) sW2[e]=W2[e];
  // un-permute: stored c' = h*64+d ; true channel c = d*4+h
  for (int e=t;e<64*256;e+=256) {
    int p = e>>8, cs = e&255;
    int c = ((cs & 63) << 2) | (cs >> 6);
    sM[p][c] = srcp[e];
  }
  __syncthreads();
  const int p = t & 63, g = t >> 6;
  for (int ii=0; ii<8; ++ii) {
    int i = g*8+ii;
    float acc = b1[i];
    const float* w = &sW1[i*256];
    #pragma unroll 8
    for (int c=0;c<256;++c) acc = fmaf(w[c], sM[p][c], acc);
    sH[p][i] = acc;
  }
  __syncthreads();
  for (int oo=0; oo<64; ++oo) {
    int o = g*64+oo;
    float acc = b2[o];
    const float* w = &sW2[o*32];
    #pragma unroll
    for (int i=0;i<32;++i) acc = fmaf(w[i], sH[p][i], acc);
    sM[p][o] = acc;                    // stage for coalesced writeback
  }
  __syncthreads();
  float* dst = m2T + ((size_t)b*N + n0)*256;
  for (int e=t;e<64*256;e+=256) dst[e] = sM[e>>8][e&255];
}

// ---------------- final mlp: concat(x, msg2) -> 64 -> BN/ReLU -> 256, out c-major
__global__ __launch_bounds__(256) void mlp_kernel(
    const float* __restrict__ X, const float* __restrict__ m2T,
    const float* __restrict__ Wp1, const float* __restrict__ bp1,
    const float* __restrict__ g1, const float* __restrict__ be1,
    const float* __restrict__ Wp2, const float* __restrict__ bp2,
    float* __restrict__ out, int N)
{
  __shared__ float sIn[512][33];
  __shared__ float sH[32][65];
  const int t = threadIdx.x;
  const int b = blockIdx.y;
  const int n0 = blockIdx.x*32;
  for (int e=t; e<256*32; e+=256) {                  // x half
    int c = e>>5, p = e&31;
    sIn[c][p] = X[((size_t)b*256 + c)*N + n0 + p];
  }
  for (int e=t; e<32*256; e+=256) {                  // msg2 half
    int p = e>>8, c = e&255;
    sIn[256+c][p] = m2T[((size_t)b*N + n0 + p)*256 + c];
  }
  __syncthreads();
  const int p = t & 31, g = t >> 5;
  const float bnsc = rsqrtf(1.0f + 1e-5f);
  for (int jj=0; jj<8; ++jj) {
    int j = g*8 + jj;
    float acc = bp1[j];
    const float* w = &Wp1[(size_t)j*512];            // uniform -> scalar loads
    #pragma unroll 8
    for (int c=0;c<512;++c) acc = fmaf(w[c], sIn[c][p], acc);
    acc = g1[j]*acc*bnsc + be1[j];
    sH[p][j] = fmaxf(acc, 0.0f);
  }
  __syncthreads();
  for (int oo=0; oo<32; ++oo) {
    int o = g*32 + oo;
    float acc = bp2[o];
    const float* w = &Wp2[(size_t)o*64];
    #pragma unroll
    for (int i=0;i<64;++i) acc = fmaf(w[i], sH[p][i], acc);
    out[((size_t)b*256 + o)*N + n0 + p] = acc;       // coalesced
  }
}

extern "C" void kernel_launch(void* const* d_in, const int* in_sizes, int n_in,
                              void* d_out, int out_size, void* d_ws, size_t ws_size,
                              hipStream_t stream) {
  (void)in_sizes; (void)n_in; (void)out_size; (void)ws_size;
  const float* x   = (const float*)d_in[0];
  const float* src = (const float*)d_in[1];
  const float* Wq1=(const float*)d_in[2],  *bq1=(const float*)d_in[3];
  const float* Wq2=(const float*)d_in[4],  *bq2=(const float*)d_in[5];
  const float* Wk1=(const float*)d_in[6],  *bk1=(const float*)d_in[7];
  const float* Wk2=(const float*)d_in[8],  *bk2=(const float*)d_in[9];
  const float* Wv1=(const float*)d_in[10], *bv1=(const float*)d_in[11];
  const float* Wv2=(const float*)d_in[12], *bv2=(const float*)d_in[13];
  const float* Wm1=(const float*)d_in[14], *bm1=(const float*)d_in[15];
  const float* Wm2=(const float*)d_in[16], *bm2=(const float*)d_in[17];
  const float* Wp1=(const float*)d_in[18], *bp1=(const float*)d_in[19];
  const float* g1 =(const float*)d_in[20], *be1=(const float*)d_in[21];
  const float* Wp2=(const float*)d_in[22], *bp2=(const float*)d_in[23];

  const size_t qkvElems = (size_t)Bb*Hh*Nn*64;       // 2 Mi f16 each
  _Float16* Qt = (_Float16*)d_ws;
  _Float16* Kt = Qt + qkvElems;
  _Float16* Vm = Kt + qkvElems;
  float* msgT  = (float*)(Vm + qkvElems);            // [B,N,256] head-major channels
  float* m2T   = msgT + (size_t)Bb*Nn*Dd;            // [B,N,256]

  proj_kernel<<<dim3(Nn/64, Bb), 256, 0, stream>>>(x,   Wq1,bq1,Wq2,bq2, Qt, Nn, 0);
  proj_kernel<<<dim3(Mm/64, Bb), 256, 0, stream>>>(src, Wk1,bk1,Wk2,bk2, Kt, Mm, 1);
  proj_kernel<<<dim3(Mm/64, Bb), 256, 0, stream>>>(src, Wv1,bv1,Wv2,bv2, Vm, Mm, 2);
  attn_kernel<<<dim3(Nn/64, Hh, Bb), 128, 0, stream>>>(Qt, Kt, Vm, msgT, Nn, Mm);
  merge_kernel<<<dim3(Nn/64, Bb), 256, 0, stream>>>(msgT, Wm1,bm1,Wm2,bm2, m2T, Nn);
  mlp_kernel<<<dim3(Nn/32, Bb), 256, 0, stream>>>(x, m2T, Wp1,bp1,g1,be1,Wp2,bp2,
                                                  (float*)d_out, Nn);
}
